// MockAutoencoder_49821620633869
// MI455X (gfx1250) — matrix operand order repaired and checked by reading, not compiled, not run on hardware
//
#include <hip/hip_runtime.h>

// ---------------------------------------------------------------------------
// MockAutoencoder (encoder conv1d -> 8-stage RVQ -> decoder conv1d + loss)
// CDNA5 / gfx1250: RVQ nearest-neighbor search runs on v_wmma_f32_16x16x32_bf16
// ---------------------------------------------------------------------------

#define BATCH   8
#define TLEN    8192
#define NTOK    (BATCH * TLEN)   // 65536 tokens
#define D_DIM   256
#define K_CODES 1024
#define STAGES  8

typedef __attribute__((ext_vector_type(16))) __bf16 v16bf;
typedef __attribute__((ext_vector_type(8)))  __bf16 v8bf;
typedef __attribute__((ext_vector_type(8)))  float  v8f;

// ---------------------------------------------------------------------------
// Kernel 1: codebook prep — bf16 copy + c2n[k] = -0.5*||c_k||^2, zero loss.
// grid = STAGES*K_CODES blocks of 256 (one block per (stage, code))
// ---------------------------------------------------------------------------
__global__ __launch_bounds__(256) void prep_kernel(
    const float* __restrict__ cb, __bf16* __restrict__ cbh,
    float* __restrict__ c2n, float* __restrict__ lossAcc)
{
    __shared__ float sred[256];
    const int sk = blockIdx.x;
    const int d  = threadIdx.x;
    const size_t o = (size_t)sk * D_DIM + d;
    const float v = cb[o];
    cbh[o] = (__bf16)v;
    sred[d] = v * v;
    __syncthreads();
    for (int s = 128; s > 0; s >>= 1) {
        if (d < s) sred[d] += sred[d + s];
        __syncthreads();
    }
    if (d == 0) c2n[sk] = -0.5f * sred[0];
    if (sk == 0 && d == 0) *lossAcc = 0.0f;
}

// ---------------------------------------------------------------------------
// Kernel 2: encoder conv1d (1 -> 256 ch, k=3, SAME). Writes initial residual
// (bf16, token-major [N,D]) and zeros quant_sum.
// grid = NTOK blocks of 256 (block = token, thread = channel)
// ---------------------------------------------------------------------------
__global__ __launch_bounds__(256) void encode_kernel(
    const float* __restrict__ x, const float* __restrict__ enc_w,
    const float* __restrict__ enc_b,
    __bf16* __restrict__ rb, float* __restrict__ qs)
{
    const int n = blockIdx.x;
    const int c = threadIdx.x;
    const int b = n >> 13;          // / TLEN
    const int t = n & (TLEN - 1);
    const float* xr = x + (size_t)b * TLEN;
    float acc = enc_b[c];
#pragma unroll
    for (int kk = 0; kk < 3; ++kk) {
        const int tt = t + kk - 1;
        if (tt >= 0 && tt < TLEN) acc += enc_w[c * 3 + kk] * xr[tt];
    }
    const size_t o = (size_t)n * D_DIM + c;
    rb[o] = (__bf16)acc;
    qs[o] = 0.0f;
}

// ---------------------------------------------------------------------------
// Kernel 3: RVQ stage argmax via WMMA.
// Each wave owns 32 tokens: two 16-token residual fragment sets (128 VGPRs)
// stay resident; 64 code tiles stream through a double-buffered A operand,
// with TWO WMMAs issued per A-fragment load (2x codebook reuse vs 16-token
// version). Accumulators are seeded with -0.5*||c||^2 so the score
// r.c - 0.5*||c||^2 falls straight out of the matrix op; argmax update is
// branch-free cmp+select (ascending scan + strict '>' keeps lowest index).
// grid = NTOK/128 blocks of 128 (4 waves) — even division, EXEC all ones.
// ---------------------------------------------------------------------------
__global__ __launch_bounds__(128) void rvq_argmax_kernel(
    const __bf16* __restrict__ rb, const __bf16* __restrict__ cbh,
    const float* __restrict__ c2n, unsigned int* __restrict__ idxOut)
{
    const int lane    = threadIdx.x & 31;
    const int wave    = threadIdx.x >> 5;
    const int n0      = (blockIdx.x * 4 + wave) * 32;
    const int halfSel = lane >> 4;      // 0: lanes 0-15, 1: lanes 16-31
    const int lrow    = lane & 15;

    // B operand: residual for 2 x 16 tokens, 8 fragments each. Contiguous.
    v16bf bfragA[8], bfragB[8];
    const __bf16* rbaseA = rb + (size_t)(n0 + lrow) * D_DIM + halfSel * 16;
    const __bf16* rbaseB = rbaseA + (size_t)16 * D_DIM;
#pragma unroll
    for (int f = 0; f < 8; ++f) {
        bfragA[f] = *(const v16bf*)(rbaseA + f * 32);
        bfragB[f] = *(const v16bf*)(rbaseB + f * 32);
    }

    float bestA = -3.4e38f, bestB = -3.4e38f;
    int   bestIdxA = 0,     bestIdxB = 0;

    for (int kt = 0; kt < 64; ++kt) {
        const int cbase = kt * 16 + halfSel * 8;
        const __bf16* abase = cbh + (size_t)(kt * 16 + lrow) * D_DIM + halfSel * 16;
        if (kt < 63)  // pull next code tile into near caches (global_prefetch_b8)
            __builtin_prefetch((const void*)(abase + 16 * D_DIM), 0, 3);

        // seed both accumulators with the -0.5*||c||^2 bias (one 32B load)
        const v8f bias = *(const v8f*)(c2n + cbase);
        v8f accA = bias, accB = bias;

        // A operand: explicit double buffer -> loads for fragment f+1 fly
        // while the two WMMAs of fragment f execute.
        union AU { v16bf v; v8bf h[2]; } a[2];
        a[0].h[0] = *(const v8bf*)(abase);
        a[0].h[1] = *(const v8bf*)(abase + 8);
#pragma unroll
        for (int f = 0; f < 8; ++f) {
            if (f < 7) {
                a[(f + 1) & 1].h[0] = *(const v8bf*)(abase + (f + 1) * 32);
                a[(f + 1) & 1].h[1] = *(const v8bf*)(abase + (f + 1) * 32 + 8);
            }
            accA = __builtin_amdgcn_wmma_f32_16x16x32_bf16(
                false, a[f & 1].v, false, bfragA[f], (short)0, accA, false, false);
            accB = __builtin_amdgcn_wmma_f32_16x16x32_bf16(
                false, a[f & 1].v, false, bfragB[f], (short)0, accB, false, false);
        }

#pragma unroll
        for (int r = 0; r < 8; ++r) {
            const int  code = cbase + r;
            const bool tA = accA[r] > bestA;
            bestA    = tA ? accA[r] : bestA;
            bestIdxA = tA ? code    : bestIdxA;
            const bool tB = accB[r] > bestB;
            bestB    = tB ? accB[r] : bestB;
            bestIdxB = tB ? code    : bestIdxB;
        }
    }

    // combine the two lane halves (codes 0..7 mod 16 vs 8..15 mod 16)
    {
        const float oV = __shfl_xor(bestA, 16, 32);
        const int   oI = __shfl_xor(bestIdxA, 16, 32);
        const bool  tk = (oV > bestA) || (oV == bestA && oI < bestIdxA);
        bestA    = tk ? oV : bestA;
        bestIdxA = tk ? oI : bestIdxA;
    }
    {
        const float oV = __shfl_xor(bestB, 16, 32);
        const int   oI = __shfl_xor(bestIdxB, 16, 32);
        const bool  tk = (oV > bestB) || (oV == bestB && oI < bestIdxB);
        bestB    = tk ? oV : bestB;
        bestIdxB = tk ? oI : bestIdxB;
    }

    if (lane < 16) {
        idxOut[n0 + lane]      = (unsigned int)bestIdxA;
        idxOut[n0 + 16 + lane] = (unsigned int)bestIdxB;
    }
}

// ---------------------------------------------------------------------------
// Kernel 4: RVQ stage update — residual -= q, quant_sum += q, loss += r_new^2
// grid = NTOK/8 blocks of 256 (block = 8 tokens, thread = dim); one loss
// atomic per block.
// ---------------------------------------------------------------------------
__global__ __launch_bounds__(256) void rvq_update_kernel(
    const float* __restrict__ cb_s, const unsigned int* __restrict__ idx,
    __bf16* __restrict__ rb, float* __restrict__ qs,
    float* __restrict__ lossAcc)
{
    __shared__ float sred[256];
    const int d  = threadIdx.x;
    const int n0 = blockIdx.x * 8;
    float part = 0.0f;
#pragma unroll
    for (int i = 0; i < 8; ++i) {
        const int n = n0 + i;
        const unsigned int k = idx[n];
        const float q = cb_s[(size_t)k * D_DIM + d];
        const size_t o = (size_t)n * D_DIM + d;
        const float r = (float)rb[o] - q;
        rb[o] = (__bf16)r;
        qs[o] += q;
        part += r * r;
    }
    sred[d] = part;
    __syncthreads();
    for (int s = 128; s > 0; s >>= 1) {
        if (d < s) sred[d] += sred[d + s];
        __syncthreads();
    }
    if (d == 0) atomicAdd(lossAcc, sred[0]);
}

// ---------------------------------------------------------------------------
// Kernel 5: decoder conv1d (256 -> 1 ch, k=3, SAME) on quant_sum + loss write.
// grid = NTOK/8 blocks of 256 (wave = token), lane-parallel over dims.
// ---------------------------------------------------------------------------
__global__ __launch_bounds__(256) void decode_kernel(
    const float* __restrict__ qs, const float* __restrict__ dec_w,
    const float* __restrict__ dec_b, const float* __restrict__ lossAcc,
    float* __restrict__ out)
{
    const int lane = threadIdx.x & 31;
    const int wave = threadIdx.x >> 5;
    const int n = blockIdx.x * 8 + wave;
    const int b = n >> 13;
    const int t = n & (TLEN - 1);
    float acc = 0.0f;
#pragma unroll
    for (int kk = 0; kk < 3; ++kk) {
        const int tt = t + kk - 1;       // uniform per wave -> no divergence
        if (tt < 0 || tt >= TLEN) continue;
        const float* qrow = qs + (size_t)(b * TLEN + tt) * D_DIM;
        for (int d = lane; d < D_DIM; d += 32)
            acc += dec_w[d * 3 + kk] * qrow[d];
    }
    for (int off = 16; off > 0; off >>= 1) acc += __shfl_xor(acc, off, 32);
    if (lane == 0) out[n] = acc + dec_b[0];

    if (blockIdx.x == 0 && threadIdx.x == 0)
        out[NTOK] = lossAcc[0] * (1.0f / ((float)NTOK * (float)D_DIM));
}

// ---------------------------------------------------------------------------
// Launch: inputs are (x, enc_w, enc_b, dec_w, dec_b, codebooks), all fp32.
// Output: decoded [65536] ++ commit_loss [1].
// ---------------------------------------------------------------------------
extern "C" void kernel_launch(void* const* d_in, const int* in_sizes, int n_in,
                              void* d_out, int out_size, void* d_ws, size_t ws_size,
                              hipStream_t stream) {
    const float* x     = (const float*)d_in[0];
    const float* enc_w = (const float*)d_in[1];
    const float* enc_b = (const float*)d_in[2];
    const float* dec_w = (const float*)d_in[3];
    const float* dec_b = (const float*)d_in[4];
    const float* cb    = (const float*)d_in[5];
    float* out = (float*)d_out;

    // workspace layout (bytes)
    char* ws = (char*)d_ws;
    __bf16*       rb   = (__bf16*)(ws);                      //  32 MB residual bf16 [N,D]
    float*        qs   = (float*)(ws + 33554432ull);         //  64 MB quant_sum f32 [N,D]
    __bf16*       cbh  = (__bf16*)(ws + 100663296ull);       //   4 MB codebooks bf16
    float*        c2n  = (float*)(ws + 104857600ull);        //  32 KB -0.5*||c||^2
    unsigned int* idx  = (unsigned int*)(ws + 104890368ull); // 256 KB per-stage indices
    float*        loss = (float*)(ws + 105152512ull);        //   4 B loss accumulator

    prep_kernel<<<STAGES * K_CODES, 256, 0, stream>>>(cb, cbh, c2n, loss);
    encode_kernel<<<NTOK, 256, 0, stream>>>(x, enc_w, enc_b, rb, qs);

    for (int s = 0; s < STAGES; ++s) {
        rvq_argmax_kernel<<<NTOK / 128, 128, 0, stream>>>(
            rb, cbh + (size_t)s * K_CODES * D_DIM, c2n + s * K_CODES, idx);
        rvq_update_kernel<<<NTOK / 8, 256, 0, stream>>>(
            cb + (size_t)s * K_CODES * D_DIM, idx, rb, qs, loss);
    }

    decode_kernel<<<NTOK / 8, 256, 0, stream>>>(qs, dec_w, dec_b, loss, out);
}
